// AFNO2D_2216203124880
// MI455X (gfx1250) — matrix operand order repaired and checked by reading, not compiled, not run on hardware
//
#include <hip/hip_runtime.h>
#include <hip/hip_bf16.h>

// AFNO2D on MI455X (gfx1250): all DFTs expressed as f16 WMMA GEMMs against
// precomputed DFT bases; per-block complex MLP as batched WMMA GEMMs.
// Generic complex-GEMM: v_wmma_f32_16x16x32_f16 (wave32), 32x32 macro-tile
// = 2x2 waves x 16x16 WMMA tiles, double-buffered LDS staging with
// global_load_async_to_lds_b128 for f16 A tiles, one barrier per K-chunk,
// unguarded epilogue stores on interior tiles.

#define USE_ASYNC_LDS 1

typedef __attribute__((ext_vector_type(16))) _Float16 v16h;
typedef __attribute__((ext_vector_type(8)))  float    v8f;

union FragU { uint4 u[2]; v16h v; };
union Half8 { _Float16 h[8]; uint4 u; };

__device__ __forceinline__ float softshrink(float v) {
    const float L = 0.01f;
    return v > L ? (v - L) : (v < -L ? (v + L) : 0.0f);
}

__device__ __forceinline__ Half8 cvt8(const float* p) {
    Half8 h;
    float4 a = *(const float4*)p, b = *(const float4*)(p + 4);
    h.h[0]=(_Float16)a.x; h.h[1]=(_Float16)a.y; h.h[2]=(_Float16)a.z; h.h[3]=(_Float16)a.w;
    h.h[4]=(_Float16)b.x; h.h[5]=(_Float16)b.y; h.h[6]=(_Float16)b.z; h.h[7]=(_Float16)b.w;
    return h;
}

// One async copy: every active lane moves 16 bytes global -> LDS.
__device__ __forceinline__ void asyncCopy16(unsigned ldsOff, const void* gptr) {
#if USE_ASYNC_LDS
    asm volatile("global_load_async_to_lds_b128 %0, %1, off"
                 :: "v"(ldsOff), "v"(gptr) : "memory");
#else
    (void)ldsOff; (void)gptr;
#endif
}
__device__ __forceinline__ void asyncWait() {
#if USE_ASYNC_LDS
    asm volatile("s_wait_asynccnt 0x0" ::: "memory");
#endif
}

// Shared, single-emission guarded tile fill (edge tiles only).
// dst tile stride 40 halves. transposed=0: dst[a*40+b]; transposed=1: dst[b*40+a]
__device__ __attribute__((noinline)) void fill_slow(
    _Float16* dst, const void* src, int isF32, long long off, int ld,
    int q0, int qMax, int p0, int pMax, int transposed, int t)
{
    for (int i = t; i < 1024; i += 128) {
        int a = i >> 5, b = i & 31;
        float v = 0.0f;
        if ((q0 + a < qMax) && (p0 + b < pMax)) {
            long long gi = off + (long long)(q0 + a) * ld + (p0 + b);
            v = isF32 ? ((const float*)src)[gi] : (float)((const _Float16*)src)[gi];
        }
        dst[transposed ? (b * 40 + a) : (a * 40 + b)] = (_Float16)v;
    }
}

// Fragment gather + WMMA step for one 32-deep K chunk.
template<bool HAS_BI>
__device__ __forceinline__ void mma_step(
    const _Float16* tAr, const _Float16* tAi,
    const _Float16* tBr, const _Float16* tBi,
    int mSub, int nSub, int lane,
    v8f& p1, v8f& p2, v8f& p3, v8f& p4)
{
    // A 16x32 f16: lane(0-15)=row M, K {0..7,16..23}; lane(16-31) K {8..15,24..31}
    const int rA  = mSub + (lane & 15);
    const int kbA = (lane >> 4) * 8;
    FragU a_r, a_i, b_r, b_i;
    a_r.u[0] = *(const uint4*)&tAr[rA * 40 + kbA];
    a_r.u[1] = *(const uint4*)&tAr[rA * 40 + kbA + 16];
    a_i.u[0] = *(const uint4*)&tAi[rA * 40 + kbA];
    a_i.u[1] = *(const uint4*)&tAi[rA * 40 + kbA + 16];
    // B 32x16 f16 (transposed tile): lane(0-15)=col N, K {0..7,16..23}; lane(16-31) K {8..15,24..31}
    const int nB  = nSub + (lane & 15);
    const int kbB = (lane >> 4) * 8;
    b_r.u[0] = *(const uint4*)&tBr[nB * 40 + kbB];
    b_r.u[1] = *(const uint4*)&tBr[nB * 40 + kbB + 16];
    p1 = __builtin_amdgcn_wmma_f32_16x16x32_f16(false, a_r.v, false, b_r.v, (short)0, p1, false, false);
    p4 = __builtin_amdgcn_wmma_f32_16x16x32_f16(false, a_i.v, false, b_r.v, (short)0, p4, false, false);
    if constexpr (HAS_BI) {
        b_i.u[0] = *(const uint4*)&tBi[nB * 40 + kbB];
        b_i.u[1] = *(const uint4*)&tBi[nB * 40 + kbB + 16];
        p2 = __builtin_amdgcn_wmma_f32_16x16x32_f16(false, a_i.v, false, b_i.v, (short)0, p2, false, false);
        p3 = __builtin_amdgcn_wmma_f32_16x16x32_f16(false, a_r.v, false, b_i.v, (short)0, p3, false, false);
    }
}

// Complex GEMM: (Cr,Ci) = (Ar + i Ai)(Br + i Bi), batched by blockIdx.z.
// HAS_BI=false: B real -> Cr = Ar*B, Ci = Ai*B (2 WMMAs instead of 4).
// EP: 0 = store f16; 1 = +bias, relu, f16; 2 = +bias, softshrink, f16;
//     3 = real-only f32 store with residual add (final irfft stage).
template<bool AF32, bool BF32, int EP, bool HAS_BI>
__global__ __launch_bounds__(128) void cgemm_wmma(
    const void* __restrict__ Ar, const void* __restrict__ Ai,
    const void* __restrict__ Br, const void* __restrict__ Bi,
    void* __restrict__ Cr, void* __restrict__ Ci,
    const float* __restrict__ biasR, const float* __restrict__ biasI,
    const float* __restrict__ resid,
    int M, int N, int K, int lda, int ldb, int ldc,
    long long sA, long long sB, long long sC, int sBias)
{
    __shared__ _Float16 sAr[2][1280];   // A tiles, row-major (M x K), stride 40
    __shared__ _Float16 sAi[2][1280];
    __shared__ _Float16 sBr[2][1280];   // B tiles, TRANSPOSED (N x K), stride 40
    __shared__ _Float16 sBi[2][1280];

    const int t    = threadIdx.x;
    const int lane = t & 31;
    const int wave = t >> 5;
    const int mBase = blockIdx.x * 32;
    const int nBase = blockIdx.y * 32;
    const int z     = blockIdx.z;
    const long long aOff = (long long)z * sA;
    const long long bOff = (long long)z * sB;
    const long long cOff = (long long)z * sC;

    const int mSub = (wave & 1) * 16;
    const int nSub = (wave >> 1) * 16;

    const bool interior = ((mBase + 32) <= M) && ((nBase + 32) <= N);
    const int  rF = t >> 2;            // fast path: row(A) / k(B) index 0..31
    const int  qF = (t & 3) * 8;       // 8-element sub-chunk

    v8f p1{}, p2{}, p3{}, p4{};

    auto fill = [&](int kc, int buf) {
        const int kBase = kc << 5;
        if (interior && (kBase + 32) <= K) {
            // ---- A tiles (async for f16, cvt path for f32) ----
            const long long ga = aOff + (long long)(mBase + rF) * lda + (kBase + qF);
            if constexpr (AF32) {
                *(uint4*)&sAr[buf][rF * 40 + qF] = cvt8((const float*)Ar + ga).u;
                *(uint4*)&sAi[buf][rF * 40 + qF] = cvt8((const float*)Ai + ga).u;
            } else {
#if USE_ASYNC_LDS
                asyncCopy16((unsigned)(size_t)&sAr[buf][rF * 40 + qF], (const _Float16*)Ar + ga);
                asyncCopy16((unsigned)(size_t)&sAi[buf][rF * 40 + qF], (const _Float16*)Ai + ga);
#else
                *(uint4*)&sAr[buf][rF * 40 + qF] = *(const uint4*)((const _Float16*)Ar + ga);
                *(uint4*)&sAi[buf][rF * 40 + qF] = *(const uint4*)((const _Float16*)Ai + ga);
#endif
            }
            // ---- B tiles (vector load along N, transpose-scatter into LDS) ----
            const long long gb = bOff + (long long)(kBase + rF) * ldb + (nBase + qF);
            Half8 hr, hi;
            if constexpr (BF32) {
                hr = cvt8((const float*)Br + gb);
                if constexpr (HAS_BI) hi = cvt8((const float*)Bi + gb);
            } else {
                hr.u = *(const uint4*)((const _Float16*)Br + gb);
                if constexpr (HAS_BI) hi.u = *(const uint4*)((const _Float16*)Bi + gb);
            }
            #pragma unroll
            for (int j = 0; j < 8; ++j) {
                sBr[buf][(qF + j) * 40 + rF] = hr.h[j];
                if constexpr (HAS_BI) sBi[buf][(qF + j) * 40 + rF] = hi.h[j];
            }
            // prefetch chunk after next of B (lowers to global_prefetch_b8)
            if (kBase + 64 <= K) {
                const size_t esz = BF32 ? 4 : 2;
                __builtin_prefetch((const char*)Br + (size_t)(gb + (long long)32 * ldb) * esz, 0, 3);
            }
        } else {
            fill_slow(sAr[buf], Ar, AF32, aOff, lda, mBase, M, kBase, K, 0, t);
            fill_slow(sAi[buf], Ai, AF32, aOff, lda, mBase, M, kBase, K, 0, t);
            fill_slow(sBr[buf], Br, BF32, bOff, ldb, kBase, K, nBase, N, 1, t);
            if constexpr (HAS_BI)
                fill_slow(sBi[buf], Bi, BF32, bOff, ldb, kBase, K, nBase, N, 1, t);
        }
    };

    const int nK = (K + 31) >> 5;
    fill(0, 0);
    if constexpr (!AF32) asyncWait();
    __syncthreads();

    for (int kc = 0; kc < nK; ++kc) {
        const int cur = kc & 1;
        const bool more = (kc + 1) < nK;
        if (more) fill(kc + 1, cur ^ 1);                 // overlap next fill ...
        mma_step<HAS_BI>(sAr[cur], sAi[cur], sBr[cur], sBi[cur],
                         mSub, nSub, lane, p1, p2, p3, p4);  // ... with WMMA work
        if (more) {
            if constexpr (!AF32) asyncWait();
            __syncthreads();                             // one barrier per chunk
        }
    }

    v8f cr, ci;
    if constexpr (HAS_BI) { cr = p1 - p2; ci = p3 + p4; }
    else                  { cr = p1;      ci = p4;      }

    const int colL = nBase + nSub + (lane & 15);
    const int hi   = (lane >> 4) * 8;   // C/D layout: vgpr g -> M = g + hi, N = lane&15
    const int row0 = mBase + mSub + hi;

    if constexpr (EP == 1 || EP == 2) {
        float bR = 0.f, bI = 0.f;
        if (interior || colL < N) {
            bR = biasR[(long long)z * sBias + colL];
            bI = biasI[(long long)z * sBias + colL];
        }
        #pragma unroll
        for (int g = 0; g < 8; ++g) {
            float vr = cr[g] + bR, vi = ci[g] + bI;
            if constexpr (EP == 1) { vr = fmaxf(vr, 0.f); vi = fmaxf(vi, 0.f); }
            else                   { vr = softshrink(vr); vi = softshrink(vi); }
            cr[g] = vr; ci[g] = vi;
        }
    }

    if constexpr (EP == 3) {
        float* Co = (float*)Cr;
        const long long o0 = cOff + (long long)row0 * ldc + colL;
        if (interior) {
            #pragma unroll
            for (int g = 0; g < 8; ++g) {
                long long o = o0 + (long long)g * ldc;
                Co[o] = cr[g] + resid[o];
            }
        } else {
            #pragma unroll
            for (int g = 0; g < 8; ++g) {
                if (row0 + g < M && colL < N) {
                    long long o = o0 + (long long)g * ldc;
                    Co[o] = cr[g] + resid[o];
                }
            }
        }
    } else {
        _Float16* Crh = (_Float16*)Cr;
        _Float16* Cih = (_Float16*)Ci;
        const long long o0 = cOff + (long long)row0 * ldc + colL;
        if (interior) {
            #pragma unroll
            for (int g = 0; g < 8; ++g) {
                long long o = o0 + (long long)g * ldc;
                Crh[o] = (_Float16)cr[g];
                Cih[o] = (_Float16)ci[g];
            }
        } else {
            #pragma unroll
            for (int g = 0; g < 8; ++g) {
                if (row0 + g < M && colL < N) {
                    long long o = o0 + (long long)g * ldc;
                    Crh[o] = (_Float16)cr[g];
                    Cih[o] = (_Float16)ci[g];
                }
            }
        }
    }
}

// Precompute DFT bases (ortho norm 1/sqrt(128) per axis; irfft Hermitian
// doubling folded into G; imaginary signs folded so every stage uses
// Cr = P1 - P2, Ci = P3 + P4).
__global__ void build_basis(_Float16* fc, _Float16* fs,
                            _Float16* fhr, _Float16* fhi,
                            _Float16* ifhr, _Float16* ifhi,
                            _Float16* gr, _Float16* gi)
{
    int id = blockIdx.x * blockDim.x + threadIdx.x;
    const float s  = 0.08838834764831845f;          // 1/sqrt(128)
    const float w0 = 6.283185307179586f / 128.0f;   // 2*pi/128
    if (id < 65 * 128) {            // forward W-rfft basis (65 x 128)
        int k = id / 128, w = id % 128;
        float th = w0 * (float)((k * w) & 127);
        fc[id] = (_Float16)( s * cosf(th));
        fs[id] = (_Float16)(-s * sinf(th));
    }
    if (id < 128 * 128) {           // forward / inverse H-fft bases (128 x 128)
        int m = id / 128, h = id % 128;
        float th = w0 * (float)((m * h) & 127);
        float c = s * cosf(th), sn = s * sinf(th);
        fhr[id]  = (_Float16)c;  fhi[id]  = (_Float16)(-sn);
        ifhr[id] = (_Float16)c;  ifhi[id] = (_Float16)( sn);
    }
    if (id < 128 * 96) {            // irfft W basis (128 x 65, K-padded to 96)
        int w = id / 96, k = id % 96;
        if (k < 65) {
            float wt = (k == 0 || k == 64) ? 1.0f : 2.0f;
            float th = w0 * (float)((w * k) & 127);
            gr[id] = (_Float16)(s * wt * cosf(th));
            gi[id] = (_Float16)(s * wt * sinf(th));
        } else { gr[id] = (_Float16)0.f; gi[id] = (_Float16)0.f; }
    }
}

extern "C" void kernel_launch(void* const* d_in, const int* in_sizes, int n_in,
                              void* d_out, int out_size, void* d_ws, size_t ws_size,
                              hipStream_t stream)
{
    const float* x  = (const float*)d_in[0];   // (4, 16384, 768) f32
    const float* w1 = (const float*)d_in[1];   // (2, 8, 96, 96)  f32
    const float* b1 = (const float*)d_in[2];   // (2, 8, 96)      f32
    const float* w2 = (const float*)d_in[3];   // (2, 8, 96, 96)  f32
    const float* b2 = (const float*)d_in[4];   // (2, 8, 96)      f32
    (void)in_sizes; (void)n_in; (void)out_size; (void)ws_size;

    // workspace carve-up (f16)
    char* ws = (char*)d_ws;
    size_t off = 0;
    auto take = [&](size_t elems) {
        _Float16* p = (_Float16*)(ws + off);
        off += (elems * sizeof(_Float16) + 255) & ~(size_t)255;
        return p;
    };
    const size_t FREQ = 512ull * 49920ull;     // (B*H) x (65*768)
    _Float16* P0r = take(FREQ);  _Float16* P0i = take(FREQ);
    _Float16* P1r = take(FREQ);  _Float16* P1i = take(FREQ);
    _Float16* fc   = take(65 * 128);   _Float16* fs   = take(65 * 128);
    _Float16* fhr  = take(128 * 128);  _Float16* fhi  = take(128 * 128);
    _Float16* ifhr = take(128 * 128);  _Float16* ifhi = take(128 * 128);
    _Float16* gr   = take(128 * 96);   _Float16* gi   = take(128 * 96);

    const dim3 blk(128);
    build_basis<<<dim3(64), dim3(256), 0, stream>>>(fc, fs, fhr, fhi, ifhr, ifhi, gr, gi);

    // S1: W-rfft.  Xr+iXi (65x768) = (Fc+iFs)(65x128) * x (128x768), batch B*H=512
    cgemm_wmma<false, true, 0, false><<<dim3(3, 24, 512), blk, 0, stream>>>(
        fc, fs, x, nullptr, P0r, P0i, nullptr, nullptr, nullptr,
        65, 768, 128, 128, 768, 768, 0, 98304, 49920, 0);

    // S2: H-fft.  (128x49920) = Fh (128x128) * X, batch B=4, full complex
    cgemm_wmma<false, false, 0, true><<<dim3(4, 1560, 4), blk, 0, stream>>>(
        fhr, fhi, P0r, P0i, P1r, P1i, nullptr, nullptr, nullptr,
        128, 49920, 128, 128, 49920, 49920, 0, 6389760, 6389760, 0);

    // L1: block MLP layer 1, batch = 8 blocks, epilogue bias+relu
    cgemm_wmma<false, true, 1, true><<<dim3(1040, 3, 8), blk, 0, stream>>>(
        P1r, P1i, w1, w1 + 8 * 96 * 96, P0r, P0i, b1, b1 + 8 * 96, nullptr,
        33280, 96, 96, 768, 96, 768, 96, 9216, 96, 96);

    // L2: block MLP layer 2, epilogue bias+softshrink
    cgemm_wmma<false, true, 2, true><<<dim3(1040, 3, 8), blk, 0, stream>>>(
        P0r, P0i, w2, w2 + 8 * 96 * 96, P1r, P1i, b2, b2 + 8 * 96, nullptr,
        33280, 96, 96, 768, 96, 768, 96, 9216, 96, 96);

    // S4: inverse H-fft (conjugate basis)
    cgemm_wmma<false, false, 0, true><<<dim3(4, 1560, 4), blk, 0, stream>>>(
        ifhr, ifhi, P1r, P1i, P0r, P0i, nullptr, nullptr, nullptr,
        128, 49920, 128, 128, 49920, 49920, 0, 6389760, 6389760, 0);

    // S5: irfft along W + residual, real f32 output
    cgemm_wmma<false, false, 3, true><<<dim3(4, 24, 512), blk, 0, stream>>>(
        gr, gi, P0r, P0i, (float*)d_out, nullptr, nullptr, nullptr, x,
        128, 768, 65, 96, 768, 768, 0, 49920, 98304, 0);
}
